// Seq2Seq_33535104648015
// MI455X (gfx1250) — compile-verified, matrix operand order checked by reading, not run on hardware
//
#include <hip/hip_runtime.h>
#include <math.h>

// Sizes (fixed by the reference)
#define V 32000
#define H 1024
#define Bv 32
#define Sv 64
#define Tv 64

typedef __attribute__((ext_vector_type(16))) __bf16 v16bf;
typedef __attribute__((ext_vector_type(8)))  float  v8f;

union FragU { uint4 q[2]; v16bf v; };

static __device__ __forceinline__ unsigned short f2bf(float f) {
    unsigned int u = __float_as_uint(f);
    unsigned int r = u + 0x7FFFu + ((u >> 16) & 1u);   // round-to-nearest-even
    return (unsigned short)(r >> 16);
}

// Load a 16x32 bf16 WMMA fragment (A layout == B layout on CDNA5):
// lane<16: rows 0..15, K {0..7} and {16..23}; lane>=16: K {8..15} and {24..31}.
// base points at element [row0, k0] of a row-major [rows, ld] bf16 matrix.
static __device__ __forceinline__ v16bf load_frag(const unsigned short* __restrict__ base,
                                                  long ld, int lane) {
    int half = lane >> 4, idx = lane & 15;
    const unsigned short* p = base + (long)idx * ld + half * 8;
    FragU f;
    f.q[0] = *reinterpret_cast<const uint4*>(p);        // K +0..7  (rel. half)
    f.q[1] = *reinterpret_cast<const uint4*>(p + 16);   // K +16..23
    return f.v;
}

static __device__ __forceinline__ v8f wmma_bf16(v16bf a, v16bf b, v8f c) {
    return __builtin_amdgcn_wmma_f32_16x16x32_bf16(false, a, false, b, (short)0, c, false, false);
}

// ---------------- elementwise helpers ----------------

__global__ void cvt_bf16(const float* __restrict__ s, unsigned short* __restrict__ d, long n) {
    long i = (long)blockIdx.x * blockDim.x + threadIdx.x;
    long stride = (long)gridDim.x * blockDim.x;
    for (; i < n; i += stride) d[i] = f2bf(s[i]);
}

__global__ void embed_enc_k(const int* __restrict__ seqs, const float* __restrict__ emb,
                            unsigned short* __restrict__ x) {
    long n = (long)Bv * Sv * H;
    long i = (long)blockIdx.x * blockDim.x + threadIdx.x;
    long stride = (long)gridDim.x * blockDim.x;
    for (; i < n; i += stride) {
        int r = (int)(i >> 10), h = (int)(i & (H - 1));
        x[i] = f2bf(emb[(long)seqs[r] * H + h]);
    }
}

// rnn_in[r=b*T+t, c] = relu(c<H ? emb_dec[tgt[b,t],c] : thought[b, c-H]) as bf16
__global__ void build_rnn_in_k(const int* __restrict__ tgt, const float* __restrict__ emb,
                               const float* __restrict__ thought, unsigned short* __restrict__ x) {
    long n = (long)Bv * Tv * 2 * H;
    long i = (long)blockIdx.x * blockDim.x + threadIdx.x;
    long stride = (long)gridDim.x * blockDim.x;
    for (; i < n; i += stride) {
        int r = (int)(i >> 11), c = (int)(i & (2 * H - 1));
        int b = r >> 6, t = r & (Tv - 1);
        float v;
        if (c < H) v = emb[(long)tgt[b * Tv + t] * H + c];
        else       v = thought[(long)b * H + (c - H)];
        v = v > 0.f ? v : 0.f;
        x[i] = f2bf(v);
    }
}

__global__ void zero_h_k(float* __restrict__ hf, unsigned short* __restrict__ hb) {
    int i = blockIdx.x * blockDim.x + threadIdx.x;
    if (i < Bv * H) { hf[i] = 0.f; hb[i] = 0; }
}

// ---------------- generic GEMM: C[M,N] = A[M,K] * B[N,K]^T + bias ----------------
// block tile 64(M) x 128(N), 8 waves, wave tile 32x32, bf16 WMMA f32 accum.
__global__ __launch_bounds__(256) void gemm_bt_bias(
    const unsigned short* __restrict__ A, const unsigned short* __restrict__ B,
    const float* __restrict__ bias, float* __restrict__ C, int N, int K) {
    const int lane = threadIdx.x & 31;
    const int wave = threadIdx.x >> 5;
    const int wm = wave & 1, wn = wave >> 1;
    const int m0 = blockIdx.y * 64 + wm * 32;
    const int n0 = blockIdx.x * 128 + wn * 32;
    v8f a00 = {}, a01 = {}, a10 = {}, a11 = {};
    const unsigned short* Ap0 = A + (long)m0 * K;
    const unsigned short* Ap1 = A + (long)(m0 + 16) * K;
    const unsigned short* Bp0 = B + (long)n0 * K;
    const unsigned short* Bp1 = B + (long)(n0 + 16) * K;
    for (int k = 0; k < K; k += 32) {
        v16bf fa0 = load_frag(Ap0 + k, K, lane);
        v16bf fa1 = load_frag(Ap1 + k, K, lane);
        v16bf fb0 = load_frag(Bp0 + k, K, lane);
        v16bf fb1 = load_frag(Bp1 + k, K, lane);
        a00 = wmma_bf16(fa0, fb0, a00);
        a01 = wmma_bf16(fa0, fb1, a01);
        a10 = wmma_bf16(fa1, fb0, a10);
        a11 = wmma_bf16(fa1, fb1, a11);
    }
    const int ml = (lane >> 4) * 8, nl = lane & 15;
    const float bv0 = bias[n0 + nl], bv1 = bias[n0 + 16 + nl];
#pragma unroll
    for (int i = 0; i < 8; ++i) {
        long r0 = (long)(m0 + ml + i) * N;
        long r1 = (long)(m0 + 16 + ml + i) * N;
        C[r0 + n0 + nl]      = a00[i] + bv0;
        C[r0 + n0 + 16 + nl] = a01[i] + bv1;
        C[r1 + n0 + nl]      = a10[i] + bv0;
        C[r1 + n0 + 16 + nl] = a11[i] + bv1;
    }
}

// ---------------- fused GRU step ----------------
// gh = h_b @ Whh^T (+b_hh), gates from xp[:,t,:], torch order [r|z|n].
// Block: 256 thr, covers all 32 batch rows x 64 hidden cols; grid = H/64.
template <bool ENC>
__global__ __launch_bounds__(256) void gru_step(
    const float* __restrict__ xp, const unsigned short* __restrict__ Whh,
    const float* __restrict__ bhh,
    const float* __restrict__ h_in, const unsigned short* __restrict__ h_in_b,
    float* __restrict__ h_out, unsigned short* __restrict__ h_out_b,
    const int* __restrict__ lens, unsigned short* __restrict__ hs_out, int t) {
    const int lane = threadIdx.x & 31;
    const int wave = threadIdx.x >> 5;
    const int wm = wave & 1, wn = wave >> 1;
    const int m0 = wm * 16;
    const int c0 = blockIdx.x * 64 + wn * 16;
    v8f ar = {}, az = {}, an = {};
    const unsigned short* Ap = h_in_b + (long)m0 * H;
    const unsigned short* Br = Whh + (long)(0 * H + c0) * H;
    const unsigned short* Bz = Whh + (long)(1 * H + c0) * H;
    const unsigned short* Bn = Whh + (long)(2 * H + c0) * H;
    for (int k = 0; k < H; k += 32) {
        v16bf fa = load_frag(Ap + k, H, lane);
        v16bf fr = load_frag(Br + k, H, lane);
        v16bf fz = load_frag(Bz + k, H, lane);
        v16bf fn = load_frag(Bn + k, H, lane);
        ar = wmma_bf16(fa, fr, ar);
        az = wmma_bf16(fa, fz, az);
        an = wmma_bf16(fa, fn, an);
    }
    const int ml = (lane >> 4) * 8, nl = lane & 15;
    const int j = c0 + nl;
    const float bhr = bhh[j], bhz = bhh[H + j], bhn = bhh[2 * H + j];
#pragma unroll
    for (int i = 0; i < 8; ++i) {
        int m = m0 + ml + i;                                   // batch row
        long xb = ((long)m * (ENC ? Sv : Tv) + t) * (3 * H);
        float xr = xp[xb + j], xz = xp[xb + H + j], xn = xp[xb + 2 * H + j];
        float rg = 1.f / (1.f + expf(-(xr + ar[i] + bhr)));
        float zg = 1.f / (1.f + expf(-(xz + az[i] + bhz)));
        float ng = tanhf(xn + rg * (an[i] + bhn));
        float hold = h_in[(long)m * H + j];
        float hnew = (1.f - zg) * ng + zg * hold;
        if (ENC) { if (t >= lens[m]) hnew = hold; }            // packed-seq freeze
        h_out[(long)m * H + j] = hnew;
        unsigned short hb = f2bf(hnew);
        h_out_b[(long)m * H + j] = hb;
        if (!ENC) hs_out[(long)m * H + j] = hb;
    }
}

// ---------------- logits GEMM + log-softmax partials ----------------
// A = hs_b [2048 x H] (row r = t*B + b), B = W_out_b [V x H], bias b_out.
// Writes logits f32 to out[b, t, v] and per-(row, n-block) (max, sumexp) partials.
#define NBLK 250
__global__ __launch_bounds__(256) void gemm_logits(
    const unsigned short* __restrict__ A, const unsigned short* __restrict__ B,
    const float* __restrict__ bias, float* __restrict__ out,
    float* __restrict__ m_part, float* __restrict__ s_part) {
    __shared__ float lm[4][64];
    __shared__ float ls[4][64];
    const int K = H, N = V;
    const int lane = threadIdx.x & 31;
    const int wave = threadIdx.x >> 5;
    const int wm = wave & 1, wn = wave >> 1;
    const int m0 = blockIdx.y * 64 + wm * 32;
    const int n0 = blockIdx.x * 128 + wn * 32;
    v8f a00 = {}, a01 = {}, a10 = {}, a11 = {};
    const unsigned short* Ap0 = A + (long)m0 * K;
    const unsigned short* Ap1 = A + (long)(m0 + 16) * K;
    const unsigned short* Bp0 = B + (long)n0 * K;
    const unsigned short* Bp1 = B + (long)(n0 + 16) * K;
    for (int k = 0; k < K; k += 32) {
        v16bf fa0 = load_frag(Ap0 + k, K, lane);
        v16bf fa1 = load_frag(Ap1 + k, K, lane);
        v16bf fb0 = load_frag(Bp0 + k, K, lane);
        v16bf fb1 = load_frag(Bp1 + k, K, lane);
        a00 = wmma_bf16(fa0, fb0, a00);
        a01 = wmma_bf16(fa0, fb1, a01);
        a10 = wmma_bf16(fa1, fb0, a10);
        a11 = wmma_bf16(fa1, fb1, a11);
    }
    const int half = lane >> 4, ml = half * 8, nl = lane & 15;
    const float bv0 = bias[n0 + nl], bv1 = bias[n0 + 16 + nl];
#pragma unroll
    for (int im = 0; im < 2; ++im) {
        float M8[8], S8[8];
#pragma unroll
        for (int i = 0; i < 8; ++i) {
            int r = m0 + im * 16 + ml + i;                     // hs row = t*B + b
            int bb = r & (Bv - 1), tt = r >> 5;
            long orow = ((long)bb * Tv + tt) * N;              // out[b, t, :]
            float v0 = (im ? a10[i] : a00[i]) + bv0;
            float v1 = (im ? a11[i] : a01[i]) + bv1;
            out[orow + n0 + nl]      = v0;
            out[orow + n0 + 16 + nl] = v1;
            float mx = fmaxf(v0, v1);
#pragma unroll
            for (int d = 1; d < 16; d <<= 1) mx = fmaxf(mx, __shfl_xor(mx, d, 32));
            float s = expf(v0 - mx) + expf(v1 - mx);
#pragma unroll
            for (int d = 1; d < 16; d <<= 1) s += __shfl_xor(s, d, 32);
            M8[i] = mx; S8[i] = s;
        }
        if (nl == 0) {
#pragma unroll
            for (int i = 0; i < 8; ++i) {
                int rb = wm * 32 + im * 16 + ml + i;
                lm[wn][rb] = M8[i];
                ls[wn][rb] = S8[i];
            }
        }
    }
    __syncthreads();
    if (threadIdx.x < 64) {
        float M = lm[0][threadIdx.x];
        for (int w = 1; w < 4; ++w) M = fmaxf(M, lm[w][threadIdx.x]);
        float S = 0.f;
        for (int w = 0; w < 4; ++w) S += ls[w][threadIdx.x] * expf(lm[w][threadIdx.x] - M);
        int rg = blockIdx.y * 64 + threadIdx.x;
        m_part[(long)rg * NBLK + blockIdx.x] = M;
        s_part[(long)rg * NBLK + blockIdx.x] = S;
    }
}

__global__ void lse_reduce(const float* __restrict__ m_part, const float* __restrict__ s_part,
                           float* __restrict__ lse) {
    int r = blockIdx.x * blockDim.x + threadIdx.x;
    if (r >= Bv * Tv) return;
    float M = -3.4e38f;
    for (int i = 0; i < NBLK; ++i) M = fmaxf(M, m_part[(long)r * NBLK + i]);
    float S = 0.f;
    for (int i = 0; i < NBLK; ++i) S += s_part[(long)r * NBLK + i] * expf(m_part[(long)r * NBLK + i] - M);
    lse[r] = M + logf(S);
}

__global__ void sub_lse(float* __restrict__ out, const float* __restrict__ lse) {
    int orow = blockIdx.x;                 // b*T + t
    int b = orow >> 6, t = orow & (Tv - 1);
    float l = lse[t * Bv + b];             // lse indexed by hs row t*B+b
    long base = (long)orow * V;
    for (int v = threadIdx.x; v < V; v += blockDim.x) out[base + v] -= l;
}

// ---------------- host launcher ----------------
extern "C" void kernel_launch(void* const* d_in, const int* in_sizes, int n_in,
                              void* d_out, int out_size, void* d_ws, size_t ws_size,
                              hipStream_t stream) {
    (void)in_sizes; (void)n_in; (void)out_size; (void)ws_size;
    const int*   input_seqs  = (const int*)d_in[0];
    const int*   target_seqs = (const int*)d_in[1];
    const int*   input_lens  = (const int*)d_in[2];
    const float* emb_enc     = (const float*)d_in[3];
    const float* W_ih_e      = (const float*)d_in[4];
    const float* W_hh_e      = (const float*)d_in[5];
    const float* b_ih_e      = (const float*)d_in[6];
    const float* b_hh_e      = (const float*)d_in[7];
    const float* emb_dec     = (const float*)d_in[8];
    const float* W_ih_d      = (const float*)d_in[9];
    const float* W_hh_d      = (const float*)d_in[10];
    const float* b_ih_d      = (const float*)d_in[11];
    const float* b_hh_d      = (const float*)d_in[12];
    const float* W_out       = (const float*)d_in[13];
    const float* b_out       = (const float*)d_in[14];
    float* out = (float*)d_out;

    // ---- workspace carve-out (~139 MB total; all offsets 256B aligned) ----
    char* ws = (char*)d_ws;
    size_t off = 0;
    auto take = [&](size_t bytes) -> void* {
        void* p = ws + off;
        off += (bytes + 255) & ~(size_t)255;
        return p;
    };
    unsigned short* wb_ih_e = (unsigned short*)take((size_t)3 * H * H * 2);
    unsigned short* wb_hh_e = (unsigned short*)take((size_t)3 * H * H * 2);
    unsigned short* wb_ih_d = (unsigned short*)take((size_t)3 * H * 2 * H * 2);
    unsigned short* wb_hh_d = (unsigned short*)take((size_t)3 * H * H * 2);
    unsigned short* wb_out  = (unsigned short*)take((size_t)V * H * 2);
    // phase-shared region: enc {x_enc bf16, xp_e f32} then dec {rnn_in bf16, xp_d f32}
    char* phase = (char*)take(33554432);
    unsigned short* x_enc  = (unsigned short*)phase;                       // 2048x1024 bf16
    float*          xp_e   = (float*)(phase + (size_t)Bv * Sv * H * 2);    // 2048x3072 f32
    unsigned short* rnn_in = (unsigned short*)phase;                       // 2048x2048 bf16
    float*          xp_d   = (float*)(phase + (size_t)Bv * Tv * 2 * H * 2);// 2048x3072 f32
    float*          h_f[2]; h_f[0] = (float*)take((size_t)Bv * H * 4);
    h_f[1] = (float*)take((size_t)Bv * H * 4);
    unsigned short* h_b[2]; h_b[0] = (unsigned short*)take((size_t)Bv * H * 2);
    h_b[1] = (unsigned short*)take((size_t)Bv * H * 2);
    unsigned short* hs_b   = (unsigned short*)take((size_t)Tv * Bv * H * 2);
    float* m_part = (float*)take((size_t)Bv * Tv * NBLK * 4);
    float* s_part = (float*)take((size_t)Bv * Tv * NBLK * 4);
    float* lse    = (float*)take((size_t)Bv * Tv * 4);

    dim3 thr(256);
    auto cvt = [&](const float* s, unsigned short* d, long n) {
        int blocks = (int)((n + 255) / 256); if (blocks > 4096) blocks = 4096;
        cvt_bf16<<<blocks, thr, 0, stream>>>(s, d, n);
    };

    // weight conversions (f32 -> bf16, once per call)
    cvt(W_ih_e, wb_ih_e, (long)3 * H * H);
    cvt(W_hh_e, wb_hh_e, (long)3 * H * H);
    cvt(W_ih_d, wb_ih_d, (long)3 * H * 2 * H);
    cvt(W_hh_d, wb_hh_d, (long)3 * H * H);
    cvt(W_out,  wb_out,  (long)V * H);

    // ---- encoder ----
    embed_enc_k<<<8192, thr, 0, stream>>>(input_seqs, emb_enc, x_enc);
    gemm_bt_bias<<<dim3(3 * H / 128, Bv * Sv / 64), thr, 0, stream>>>(
        x_enc, wb_ih_e, b_ih_e, xp_e, 3 * H, H);
    zero_h_k<<<(Bv * H + 255) / 256, thr, 0, stream>>>(h_f[0], h_b[0]);
    for (int t = 0; t < Sv; ++t) {
        int in = t & 1, o = (t + 1) & 1;
        gru_step<true><<<H / 64, thr, 0, stream>>>(
            xp_e, wb_hh_e, b_hh_e, h_f[in], h_b[in], h_f[o], h_b[o],
            input_lens, (unsigned short*)nullptr, t);
    }
    // thought = h_f[0] after 64 steps

    // ---- decoder ----
    build_rnn_in_k<<<8192, thr, 0, stream>>>(target_seqs, emb_dec, h_f[0], rnn_in);
    gemm_bt_bias<<<dim3(3 * H / 128, Bv * Tv / 64), thr, 0, stream>>>(
        rnn_in, wb_ih_d, b_ih_d, xp_d, 3 * H, 2 * H);
    zero_h_k<<<(Bv * H + 255) / 256, thr, 0, stream>>>(h_f[0], h_b[0]);
    for (int t = 0; t < Tv; ++t) {
        int in = t & 1, o = (t + 1) & 1;
        gru_step<false><<<H / 64, thr, 0, stream>>>(
            xp_d, wb_hh_d, b_hh_d, h_f[in], h_b[in], h_f[o], h_b[o],
            (const int*)nullptr, hs_b + (size_t)t * Bv * H, t);
    }

    // ---- logits + log_softmax ----
    gemm_logits<<<dim3(NBLK, Bv * Tv / 64), thr, 0, stream>>>(
        hs_b, wb_out, b_out, out, m_part, s_part);
    lse_reduce<<<(Bv * Tv + 255) / 256, thr, 0, stream>>>(m_part, s_part, lse);
    sub_lse<<<Bv * Tv, thr, 0, stream>>>(out, lse);
}
